// QincoQuantize_19859928777282
// MI455X (gfx1250) — matrix operand order repaired and checked by reading, not compiled, not run on hardware
//
#include <hip/hip_runtime.h>
#include <hip/hip_bf16.h>

typedef __attribute__((ext_vector_type(16))) __bf16 v16bf;
typedef __attribute__((ext_vector_type(8)))  __bf16 v8bf;
typedef __attribute__((ext_vector_type(8)))  float  v8f;

#define T_TOK 8192          // B*S
#define D_MOD 768
#define D_H2  512           // D2
#define N_E   1024
#define S_LEN 1024
#define N_HEAD 8
#define HD_   96
#define GS_BLOCKS 2048

__device__ __forceinline__ __bf16 f2bf(float f) {
  unsigned u = __float_as_uint(f);
  u += 0x7FFFu + ((u >> 16) & 1u);           // round-to-nearest-even
  unsigned short h = (unsigned short)(u >> 16);
  return __builtin_bit_cast(__bf16, h);
}

// gfx1250 async copy: global -> LDS, 16B per lane, tracked by ASYNCcnt
__device__ __forceinline__ void async_ld_b128(unsigned ldsOff, unsigned voffBytes,
                                              unsigned long long sbase) {
  asm volatile("global_load_async_to_lds_b128 %0, %1, %2"
               :: "v"(ldsOff), "v"(voffBytes), "s"(sbase)
               : "memory");
}
__device__ __forceinline__ void wait_async0() {
  asm volatile("s_wait_asynccnt 0x0" ::: "memory");
}

__global__ void cast_bf16_k(const float* __restrict__ X, __bf16* __restrict__ Y, size_t n)
{
  size_t i = (size_t)blockIdx.x * blockDim.x + threadIdx.x;
  size_t stride = (size_t)gridDim.x * blockDim.x;
  for (; i < n; i += stride) Y[i] = f2bf(X[i]);
}

// ---------------------------------------------------------------------------
// WMMA GEMM: C[M,N] = A[M,K](f32) * W[N,K](bf16)^T  (+bias)(+C)(relu)
// flags: 1=add bias[col], 2=relu, 4=accumulate into existing C
// wcol: column offset into W's K dimension (for split concat-matmul)
// Double-buffered LDS: weight tiles via global_load_async_to_lds_b128,
// activation tiles prefetched into registers during WMMA compute.
// ---------------------------------------------------------------------------
__global__ __launch_bounds__(256)
void gemm_bf16_wmma(float* __restrict__ C, const float* __restrict__ A,
                    const __bf16* __restrict__ W, const float* __restrict__ bias,
                    int M, int N, int K, int lda, int ldw, int wcol, int ldc,
                    int flags)
{
  __shared__ __align__(32) __bf16 As[2][128][32];
  __shared__ __align__(32) __bf16 Bs[2][128][32];

  const int tid  = threadIdx.x;
  const int lane = tid & 31;
  const int wv   = tid >> 5;          // wave id 0..7
  const int hw   = lane >> 4;         // half-wave 0/1
  const int l16  = lane & 15;
  const int waveM = wv >> 2;          // 0..1  (64-row slabs)
  const int waveN = wv & 3;           // 0..3  (32-col slabs)
  const int rowBase = blockIdx.y * 128;
  const int colBase = blockIdx.x * 128;

  // per-thread async-copy slots for a 128x32 bf16 weight tile (8KB = 512 x b128)
  const unsigned long long wbase = (unsigned long long)(size_t)W;
  const int r0 = tid >> 1,         s0_ = (tid & 1) * 16;          // slots 0..255
  const int r1 = (tid + 256) >> 1, s1_ = (tid & 1) * 16;          // slots 256..511
  const unsigned ldsB0 = (unsigned)(size_t)&Bs[0][0][0];
  const unsigned slot0 = (unsigned)((r0 * 32 + s0_) * 2);
  const unsigned slot1 = (unsigned)((r1 * 32 + s1_) * 2);
  const unsigned gb0   = (unsigned)(((size_t)(colBase + r0) * ldw + wcol + s0_) * 2);
  const unsigned gb1   = (unsigned)(((size_t)(colBase + r1) * ldw + wcol + s1_) * 2);

  float4 pre[4];
  auto asyncB = [&](int k0, int buf) {
    unsigned base = ldsB0 + (unsigned)buf * (128 * 32 * 2);
    async_ld_b128(base + slot0, gb0 + (unsigned)(k0 << 1), wbase);
    async_ld_b128(base + slot1, gb1 + (unsigned)(k0 << 1), wbase);
  };
  auto loadA = [&](int k0) {
    #pragma unroll
    for (int it = 0; it < 4; ++it) {
      int f  = tid + 256 * it;        // 0..1023 float4 slots
      int r  = f >> 3;
      int c4 = (f & 7) << 2;
      pre[it] = *(const float4*)&A[(size_t)(rowBase + r) * lda + k0 + c4];
    }
  };
  auto storeA = [&](int buf) {
    #pragma unroll
    for (int it = 0; it < 4; ++it) {
      int f  = tid + 256 * it;
      int r  = f >> 3;
      int c4 = (f & 7) << 2;
      As[buf][r][c4 + 0] = f2bf(pre[it].x);
      As[buf][r][c4 + 1] = f2bf(pre[it].y);
      As[buf][r][c4 + 2] = f2bf(pre[it].z);
      As[buf][r][c4 + 3] = f2bf(pre[it].w);
    }
  };

  v8f acc[4][2];
  #pragma unroll
  for (int tm = 0; tm < 4; ++tm)
    #pragma unroll
    for (int tn = 0; tn < 2; ++tn)
      #pragma unroll
      for (int i = 0; i < 8; ++i) acc[tm][tn][i] = 0.f;

  // prologue: stage tile 0 into buffer 0
  asyncB(0, 0);
  loadA(0);
  storeA(0);
  wait_async0();
  __syncthreads();

  const int ntiles = K >> 5;
  for (int t = 0; t < ntiles; ++t) {
    const int cur = t & 1, nxt = cur ^ 1;
    const bool more = (t + 1) < ntiles;

    // prefetch tile t+1: DMA weights to LDS[nxt], A float4s into registers
    if (more) {
      asyncB((t + 1) * 32, nxt);
      loadA((t + 1) * 32);
    }

    // compute on tile t from LDS[cur]
    v16bf af[4], bf[2];
    #pragma unroll
    for (int tm = 0; tm < 4; ++tm) {
      int mr = waveM * 64 + tm * 16 + l16;
      v8bf lo = *(const v8bf*)&As[cur][mr][hw * 16];
      v8bf hi = *(const v8bf*)&As[cur][mr][hw * 16 + 8];
      af[tm] = __builtin_shufflevector(lo, hi, 0,1,2,3,4,5,6,7,8,9,10,11,12,13,14,15);
    }
    #pragma unroll
    for (int tn = 0; tn < 2; ++tn) {
      int nr = waveN * 32 + tn * 16 + l16;
      bf[tn] = *(const v16bf*)&Bs[cur][nr][hw * 16];    // 32B contiguous, aligned
    }
    #pragma unroll
    for (int tm = 0; tm < 4; ++tm)
      #pragma unroll
      for (int tn = 0; tn < 2; ++tn)
        acc[tm][tn] = __builtin_amdgcn_wmma_f32_16x16x32_bf16(
            false, af[tm], false, bf[tn], (short)0, acc[tm][tn], false, false);

    // commit prefetched A tile, drain DMA, single barrier per iteration
    if (more) {
      storeA(nxt);
      wait_async0();
    }
    __syncthreads();
  }

  #pragma unroll
  for (int tm = 0; tm < 4; ++tm)
    #pragma unroll
    for (int tn = 0; tn < 2; ++tn)
      #pragma unroll
      for (int i = 0; i < 8; ++i) {
        int gr = rowBase + waveM * 64 + tm * 16 + i + 8 * hw;
        int gc = colBase + waveN * 32 + tn * 16 + l16;
        size_t off = (size_t)gr * ldc + gc;
        float v = acc[tm][tn][i];
        if (flags & 1) v += bias[gc];
        if (flags & 4) v += C[off];
        if (flags & 2) v = fmaxf(v, 0.f);
        C[off] = v;
      }
}

// ---------------------------------------------------------------------------
// Flash attention: one block per (128 q rows, b*H+h). Wave owns 16 q rows.
// ---------------------------------------------------------------------------
__global__ __launch_bounds__(256)
void flash_attn_k(float* __restrict__ O, const float* __restrict__ Qg,
                  const float* __restrict__ Kg, const float* __restrict__ Vg)
{
  __shared__ __align__(32) __bf16 Ks[32][96];       // [kv][dim]
  __shared__ __align__(32) __bf16 Vt[96][32];       // [dim][kv]
  __shared__ __align__(32) __bf16 Ps[8][16][32];    // per-wave probs

  const int tid  = threadIdx.x;
  const int lane = tid & 31;
  const int wv   = tid >> 5;
  const int hw   = lane >> 4;
  const int l16  = lane & 15;
  const int bh = blockIdx.y;
  const int b = bh >> 3, h = bh & 7;
  const int qBase = blockIdx.x * 128;
  const size_t baseBH = (size_t)b * S_LEN * D_MOD + (size_t)h * HD_;

  // persistent Q fragments (16 x 96 per wave -> 3 K-chunks of 32)
  v16bf qf[3];
  {
    const float* qp = Qg + baseBH + (size_t)(qBase + wv * 16 + l16) * D_MOD;
    #pragma unroll
    for (int kc = 0; kc < 3; ++kc) {
      int d0 = kc * 32 + hw * 16;
      float4 a0 = *(const float4*)&qp[d0];
      float4 a1 = *(const float4*)&qp[d0 + 4];
      float4 b0 = *(const float4*)&qp[d0 + 8];
      float4 b1 = *(const float4*)&qp[d0 + 12];
      qf[kc][0]  = f2bf(a0.x); qf[kc][1]  = f2bf(a0.y);
      qf[kc][2]  = f2bf(a0.z); qf[kc][3]  = f2bf(a0.w);
      qf[kc][4]  = f2bf(a1.x); qf[kc][5]  = f2bf(a1.y);
      qf[kc][6]  = f2bf(a1.z); qf[kc][7]  = f2bf(a1.w);
      qf[kc][8]  = f2bf(b0.x); qf[kc][9]  = f2bf(b0.y);
      qf[kc][10] = f2bf(b0.z); qf[kc][11] = f2bf(b0.w);
      qf[kc][12] = f2bf(b1.x); qf[kc][13] = f2bf(b1.y);
      qf[kc][14] = f2bf(b1.z); qf[kc][15] = f2bf(b1.w);
    }
  }

  float mstate[8], lstate[8];
  v8f oacc[6];
  #pragma unroll
  for (int i = 0; i < 8; ++i) { mstate[i] = -3.0e38f; lstate[i] = 0.f; }
  #pragma unroll
  for (int j = 0; j < 6; ++j)
    #pragma unroll
    for (int i = 0; i < 8; ++i) oacc[j][i] = 0.f;

  const float SC = 0.10206207261596575f;            // 1/sqrt(96)

  for (int kv0 = 0; kv0 < S_LEN; kv0 += 32) {
    #pragma unroll
    for (int it = 0; it < 3; ++it) {                // 768 float4s
      int idx4 = tid + 256 * it;
      int kv = idx4 / 24, c4 = (idx4 % 24) << 2;
      float4 kk = *(const float4*)&Kg[baseBH + (size_t)(kv0 + kv) * D_MOD + c4];
      Ks[kv][c4 + 0] = f2bf(kk.x); Ks[kv][c4 + 1] = f2bf(kk.y);
      Ks[kv][c4 + 2] = f2bf(kk.z); Ks[kv][c4 + 3] = f2bf(kk.w);
      float4 vvv = *(const float4*)&Vg[baseBH + (size_t)(kv0 + kv) * D_MOD + c4];
      Vt[c4 + 0][kv] = f2bf(vvv.x); Vt[c4 + 1][kv] = f2bf(vvv.y);
      Vt[c4 + 2][kv] = f2bf(vvv.z); Vt[c4 + 3][kv] = f2bf(vvv.w);
    }
    __syncthreads();

    v8f s0, s1;
    #pragma unroll
    for (int i = 0; i < 8; ++i) { s0[i] = 0.f; s1[i] = 0.f; }
    #pragma unroll
    for (int kc = 0; kc < 3; ++kc) {
      v16bf bf0 = *(const v16bf*)&Ks[l16][kc * 32 + hw * 16];
      v16bf bf1 = *(const v16bf*)&Ks[l16 + 16][kc * 32 + hw * 16];
      s0 = __builtin_amdgcn_wmma_f32_16x16x32_bf16(false, qf[kc], false, bf0,
                                                   (short)0, s0, false, false);
      s1 = __builtin_amdgcn_wmma_f32_16x16x32_bf16(false, qf[kc], false, bf1,
                                                   (short)0, s1, false, false);
    }

    // online softmax over the 32 new columns (row = i + 8*hw, cols across 16 lanes)
    #pragma unroll
    for (int i = 0; i < 8; ++i) {
      float v0 = s0[i] * SC, v1 = s1[i] * SC;
      float mx = fmaxf(v0, v1);
      #pragma unroll
      for (int off = 8; off >= 1; off >>= 1) mx = fmaxf(mx, __shfl_xor(mx, off, 16));
      float mnew = fmaxf(mstate[i], mx);
      float p0 = __expf(v0 - mnew);
      float p1 = __expf(v1 - mnew);
      float corr = __expf(mstate[i] - mnew);
      float rs = p0 + p1;
      #pragma unroll
      for (int off = 8; off >= 1; off >>= 1) rs += __shfl_xor(rs, off, 16);
      lstate[i] = lstate[i] * corr + rs;
      mstate[i] = mnew;
      #pragma unroll
      for (int j = 0; j < 6; ++j) oacc[j][i] *= corr;
      int mrow = i + 8 * hw;
      Ps[wv][mrow][l16]      = f2bf(p0);
      Ps[wv][mrow][l16 + 16] = f2bf(p1);
    }
    __syncthreads();

    v16bf pf;
    {
      v8bf lo = *(const v8bf*)&Ps[wv][l16][hw * 16];
      v8bf hi = *(const v8bf*)&Ps[wv][l16][hw * 16 + 8];
      pf = __builtin_shufflevector(lo, hi, 0,1,2,3,4,5,6,7,8,9,10,11,12,13,14,15);
    }
    #pragma unroll
    for (int j = 0; j < 6; ++j) {
      v16bf vf = *(const v16bf*)&Vt[j * 16 + l16][hw * 16];
      oacc[j] = __builtin_amdgcn_wmma_f32_16x16x32_bf16(false, pf, false, vf,
                                                        (short)0, oacc[j], false, false);
    }
    __syncthreads();
  }

  #pragma unroll
  for (int j = 0; j < 6; ++j)
    #pragma unroll
    for (int i = 0; i < 8; ++i) {
      int gr = qBase + wv * 16 + i + 8 * hw;
      int gc = h * HD_ + j * 16 + l16;
      O[(size_t)b * S_LEN * D_MOD + (size_t)gr * D_MOD + gc] = oacc[j][i] / lstate[i];
    }
}

// ---------------------------------------------------------------------------
__global__ __launch_bounds__(256)
void colstats_k(const float* __restrict__ X, float* __restrict__ mean,
                float* __restrict__ var, int rows, int cols)
{
  __shared__ float s1[256], s2[256];
  int c = blockIdx.x;
  float a = 0.f, b = 0.f;
  for (int r = threadIdx.x; r < rows; r += 256) {
    float v = X[(size_t)r * cols + c];
    a += v; b += v * v;
  }
  s1[threadIdx.x] = a; s2[threadIdx.x] = b;
  __syncthreads();
  for (int st = 128; st > 0; st >>= 1) {
    if (threadIdx.x < st) {
      s1[threadIdx.x] += s1[threadIdx.x + st];
      s2[threadIdx.x] += s2[threadIdx.x + st];
    }
    __syncthreads();
  }
  if (threadIdx.x == 0) {
    float m = s1[0] / (float)rows;
    mean[c] = m;
    var[c] = s2[0] / (float)rows - m * m;
  }
}

__global__ void bn_relu_k(float* __restrict__ X, const float* __restrict__ mean,
                          const float* __restrict__ var, const float* __restrict__ g,
                          const float* __restrict__ bt, size_t total, int cols)
{
  size_t i = (size_t)blockIdx.x * blockDim.x + threadIdx.x;
  size_t stride = (size_t)gridDim.x * blockDim.x;
  for (; i < total; i += stride) {
    int c = (int)(i % (size_t)cols);
    float v = (X[i] - mean[c]) * rsqrtf(var[c] + 1e-5f) * g[c] + bt[c];
    X[i] = fmaxf(v, 0.f);
  }
}

__global__ __launch_bounds__(256)
void argmax_k(const float* __restrict__ Z, const float* __restrict__ G,
              int* __restrict__ Idx)
{
  __shared__ float sv[256];
  __shared__ int   si[256];
  int row = blockIdx.x;                 // m*T + t
  int t = row & (T_TOK - 1);
  const float* z = Z + (size_t)t * N_E;
  const float* g = G + (size_t)row * N_E;
  float best = -3.0e38f; int bi = 0;
  for (int e = threadIdx.x; e < N_E; e += 256) {
    float v = z[e] + g[e];
    if (v > best) { best = v; bi = e; }
  }
  sv[threadIdx.x] = best; si[threadIdx.x] = bi;
  __syncthreads();
  for (int st = 128; st > 0; st >>= 1) {
    if (threadIdx.x < st) {
      float v2 = sv[threadIdx.x + st]; int i2 = si[threadIdx.x + st];
      if (v2 > sv[threadIdx.x] || (v2 == sv[threadIdx.x] && i2 < si[threadIdx.x])) {
        sv[threadIdx.x] = v2; si[threadIdx.x] = i2;
      }
    }
    __syncthreads();
  }
  if (threadIdx.x == 0) Idx[row] = si[0];
}

__global__ void gather_k(float* __restrict__ Out, const float* __restrict__ CB,
                         const int* __restrict__ Idx)
{
  size_t total = (size_t)T_TOK * D_MOD;
  size_t i = (size_t)blockIdx.x * blockDim.x + threadIdx.x;
  size_t stride = (size_t)gridDim.x * blockDim.x;
  for (; i < total; i += stride) {
    size_t t = i / D_MOD, d = i % D_MOD;
    Out[i] = CB[(size_t)Idx[t] * D_MOD + d];
  }
}

__global__ void add_k(float* __restrict__ Y, const float* __restrict__ X, size_t total)
{
  size_t i = (size_t)blockIdx.x * blockDim.x + threadIdx.x;
  size_t stride = (size_t)gridDim.x * blockDim.x;
  for (; i < total; i += stride) Y[i] += X[i];
}

__global__ __launch_bounds__(256)
void sqerr_k(const float* __restrict__ A, const float* __restrict__ X,
             float* __restrict__ L)
{
  __shared__ float s[256];
  int t = blockIdx.x;
  float a = 0.f;
  for (int d = threadIdx.x; d < D_MOD; d += 256) {
    float df = A[(size_t)t * D_MOD + d] - X[(size_t)t * D_MOD + d];
    a += df * df;
  }
  s[threadIdx.x] = a;
  __syncthreads();
  for (int st = 128; st > 0; st >>= 1) {
    if (threadIdx.x < st) s[threadIdx.x] += s[threadIdx.x + st];
    __syncthreads();
  }
  if (threadIdx.x == 0) L[t] = s[0];
}

__global__ void rope_cache_k(float* __restrict__ cosc, float* __restrict__ sinc)
{
  int i = blockIdx.x * blockDim.x + threadIdx.x;
  if (i >= S_LEN * (HD_ / 2)) return;
  int s = i / (HD_ / 2), j = i % (HD_ / 2);
  float f = __expf(-((2.f * (float)j) / (float)HD_) * __logf(500000.f));
  float wavelen = 6.28318530717958647692f / f;
  const float oldc = 8192.f, lowf = 1.f, highf = 4.f, sf = 32.f;
  float sc;
  if (wavelen > oldc / lowf) sc = f / sf;
  else if (wavelen < oldc / highf) sc = f;
  else {
    float smooth = (oldc / wavelen - lowf) / (highf - lowf);
    sc = (1.f - smooth) * f / sf + smooth * f;
  }
  float ang = (float)s * sc;
  cosc[i] = cosf(ang);
  sinc[i] = sinf(ang);
}

__global__ void rope_apply_k(float* __restrict__ Q, const float* __restrict__ cosc,
                             const float* __restrict__ sinc)
{
  size_t total = (size_t)T_TOK * (D_MOD / 2);
  size_t i = (size_t)blockIdx.x * blockDim.x + threadIdx.x;
  size_t stride = (size_t)gridDim.x * blockDim.x;
  for (; i < total; i += stride) {
    int t = (int)(i / (D_MOD / 2));
    int p = (int)(i % (D_MOD / 2));
    int h = p / (HD_ / 2), j = p % (HD_ / 2);
    int s = t & (S_LEN - 1);
    size_t base = (size_t)t * D_MOD + h * HD_ + 2 * j;
    float c = cosc[s * (HD_ / 2) + j], sn = sinc[s * (HD_ / 2) + j];
    float x1 = Q[base], x2 = Q[base + 1];
    Q[base]     = x1 * c - x2 * sn;
    Q[base + 1] = x2 * c + x1 * sn;
  }
}

// ---------------------------------------------------------------------------
extern "C" void kernel_launch(void* const* d_in, const int* in_sizes, int n_in,
                              void* d_out, int out_size, void* d_ws, size_t ws_size,
                              hipStream_t stream)
{
  (void)in_sizes; (void)n_in; (void)out_size; (void)ws_size;
  const float* x        = (const float*)d_in[0];
  const float* gumbel   = (const float*)d_in[1];
  const float* codebook = (const float*)d_in[2];
  const float* post_w1  = (const float*)d_in[3];
  const float* post_b1  = (const float*)d_in[4];
  const float* bn1_g    = (const float*)d_in[5];
  const float* bn1_b    = (const float*)d_in[6];
  const float* post_w2  = (const float*)d_in[7];
  const float* post_b2  = (const float*)d_in[8];
  const float* bn2_g    = (const float*)d_in[9];
  const float* bn2_b    = (const float*)d_in[10];
  const float* q_w      = (const float*)d_in[11];
  const float* k_w      = (const float*)d_in[12];
  const float* v_w      = (const float*)d_in[13];
  const float* o_w      = (const float*)d_in[14];
  const float* concat_w = (const float*)d_in[15];
  const float* concat_b = (const float*)d_in[16];
  const float* res_w1   = (const float*)d_in[17];
  const float* res_w2   = (const float*)d_in[18];
  const float* dec_w1   = (const float*)d_in[19];
  const float* dec_b1   = (const float*)d_in[20];
  const float* dbn1_g   = (const float*)d_in[21];
  const float* dbn1_b   = (const float*)d_in[22];
  const float* dec_w2   = (const float*)d_in[23];
  const float* dec_b2   = (const float*)d_in[24];
  const float* dbn2_g   = (const float*)d_in[25];
  const float* dbn2_b   = (const float*)d_in[26];

  const size_t T = T_TOK;
  float* out      = (float*)d_out;                          // (B,S,D)
  int*   idx_out  = (int*)(out + T * D_MOD);                // (M,B,S) int32 bits
  float* loss_out = out + T * D_MOD + 4 * T;                // (M,B,S)

  float* ws = (float*)d_ws;
  float* p_xhat = ws;
  float* p_zqs  = ws + (size_t)768  * T;
  float* p_q    = ws + (size_t)1536 * T;
  float* p_k    = ws + (size_t)2304 * T;
  float* p_v    = ws + (size_t)3072 * T;
  float* p_attn = ws + (size_t)3840 * T;
  float* p_mean = ws + (size_t)4608 * T;
  float* p_var  = p_mean + 1536;
  float* p_cos  = p_var  + 1536;
  float* p_sin  = p_cos  + S_LEN * (HD_ / 2);
  // aliases over dead f32 buffers
  float* p_z1 = p_q;     // T x 512
  float* p_z2 = p_k;     // T x 1024 (spans into p_v region; both free then)
  float* p_h  = p_q;     // T x 1536 (spans q+k)
  float* p_a  = p_q;     // T x 768
  float* p_t  = p_k;     // T x 768

  // bf16 weight shadows (16B-aligned region after f32 scratch)
  __bf16* wb      = (__bf16*)(p_sin + S_LEN * (HD_ / 2));
  __bf16* wb_post1 = wb;                               // 512*768
  __bf16* wb_post2 = wb_post1 + (size_t)D_H2 * D_MOD;  // 1024*512
  __bf16* wb_q     = wb_post2 + (size_t)N_E * D_H2;    // 3*768*768
  __bf16* wb_k     = wb_q  + (size_t)3 * D_MOD * D_MOD;
  __bf16* wb_v     = wb_k  + (size_t)3 * D_MOD * D_MOD;
  __bf16* wb_o     = wb_v  + (size_t)3 * D_MOD * D_MOD;
  __bf16* wb_cc    = wb_o  + (size_t)3 * D_MOD * D_MOD;      // 3*768*1536
  __bf16* wb_r1    = wb_cc + (size_t)3 * D_MOD * 2 * D_MOD;  // 6*768*768
  __bf16* wb_r2    = wb_r1 + (size_t)6 * D_MOD * D_MOD;
  __bf16* wb_d1    = wb_r2 + (size_t)6 * D_MOD * D_MOD;      // 1536*768
  __bf16* wb_d2    = wb_d1 + (size_t)2 * D_MOD * D_MOD;      // 768*1536

  auto CAST = [&](const float* src, __bf16* dst, size_t n) {
    cast_bf16_k<<<1024, 256, 0, stream>>>(src, dst, n);
  };
  CAST(post_w1, wb_post1, (size_t)D_H2 * D_MOD);
  CAST(post_w2, wb_post2, (size_t)N_E * D_H2);
  CAST(q_w, wb_q, (size_t)3 * D_MOD * D_MOD);
  CAST(k_w, wb_k, (size_t)3 * D_MOD * D_MOD);
  CAST(v_w, wb_v, (size_t)3 * D_MOD * D_MOD);
  CAST(o_w, wb_o, (size_t)3 * D_MOD * D_MOD);
  CAST(concat_w, wb_cc, (size_t)3 * D_MOD * 2 * D_MOD);
  CAST(res_w1, wb_r1, (size_t)6 * D_MOD * D_MOD);
  CAST(res_w2, wb_r2, (size_t)6 * D_MOD * D_MOD);
  CAST(dec_w1, wb_d1, (size_t)2 * D_MOD * D_MOD);
  CAST(dec_w2, wb_d2, (size_t)2 * D_MOD * D_MOD);

  auto GEMM = [&](float* C, const float* A, const __bf16* W, const float* bias,
                  int M, int N, int K, int lda, int ldw, int wcol, int ldc, int flags) {
    dim3 g(N / 128, M / 128);
    gemm_bf16_wmma<<<g, 256, 0, stream>>>(C, A, W, bias, M, N, K, lda, ldw, wcol, ldc, flags);
  };

  rope_cache_k<<<(S_LEN * (HD_ / 2) + 255) / 256, 256, 0, stream>>>(p_cos, p_sin);

  // ---- encoder
  GEMM(p_z1, x, wb_post1, post_b1, T_TOK, D_H2, D_MOD, D_MOD, D_MOD, 0, D_H2, 1);
  colstats_k<<<D_H2, 256, 0, stream>>>(p_z1, p_mean, p_var, T_TOK, D_H2);
  bn_relu_k<<<GS_BLOCKS, 256, 0, stream>>>(p_z1, p_mean, p_var, bn1_g, bn1_b, T * D_H2, D_H2);
  GEMM(p_z2, p_z1, wb_post2, post_b2, T_TOK, N_E, D_H2, D_H2, D_H2, 0, N_E, 1);
  colstats_k<<<N_E, 256, 0, stream>>>(p_z2, p_mean, p_var, T_TOK, N_E);
  bn_relu_k<<<GS_BLOCKS, 256, 0, stream>>>(p_z2, p_mean, p_var, bn2_g, bn2_b, T * N_E, N_E);

  // ---- gumbel argmax, stage-0 reconstruction + loss
  argmax_k<<<4 * T_TOK, 256, 0, stream>>>(p_z2, gumbel, idx_out);
  gather_k<<<GS_BLOCKS, 256, 0, stream>>>(p_xhat, codebook, idx_out);
  sqerr_k<<<T_TOK, 256, 0, stream>>>(p_xhat, x, loss_out);

  // ---- residual stages
  for (int i = 1; i < 4; ++i) {
    int wi = i - 1;
    gather_k<<<GS_BLOCKS, 256, 0, stream>>>(
        p_zqs, codebook + (size_t)i * N_E * D_MOD, idx_out + (size_t)i * T);

    GEMM(p_q, p_xhat, wb_q + (size_t)wi * D_MOD * D_MOD, nullptr,
         T_TOK, D_MOD, D_MOD, D_MOD, D_MOD, 0, D_MOD, 0);
    GEMM(p_k, p_xhat, wb_k + (size_t)wi * D_MOD * D_MOD, nullptr,
         T_TOK, D_MOD, D_MOD, D_MOD, D_MOD, 0, D_MOD, 0);
    GEMM(p_v, p_xhat, wb_v + (size_t)wi * D_MOD * D_MOD, nullptr,
         T_TOK, D_MOD, D_MOD, D_MOD, D_MOD, 0, D_MOD, 0);
    rope_apply_k<<<GS_BLOCKS, 256, 0, stream>>>(p_q, p_cos, p_sin);
    rope_apply_k<<<GS_BLOCKS, 256, 0, stream>>>(p_k, p_cos, p_sin);

    flash_attn_k<<<dim3(S_LEN / 128, 8 * N_HEAD), 256, 0, stream>>>(p_attn, p_q, p_k, p_v);

    GEMM(p_a, p_attn, wb_o + (size_t)wi * D_MOD * D_MOD, nullptr,
         T_TOK, D_MOD, D_MOD, D_MOD, D_MOD, 0, D_MOD, 0);

    // zqs += concat([zqs, a]) @ Wc.T + bc  (split-K, no concat materialized)
    GEMM(p_t, p_zqs, wb_cc + (size_t)wi * D_MOD * 2 * D_MOD, concat_b + (size_t)wi * D_MOD,
         T_TOK, D_MOD, D_MOD, D_MOD, 2 * D_MOD, 0, D_MOD, 1);
    GEMM(p_t, p_a, wb_cc + (size_t)wi * D_MOD * 2 * D_MOD, nullptr,
         T_TOK, D_MOD, D_MOD, D_MOD, 2 * D_MOD, D_MOD, D_MOD, 4);
    add_k<<<GS_BLOCKS, 256, 0, stream>>>(p_zqs, p_t, T * D_MOD);

    for (int r = 0; r < 6; ++r) {
      GEMM(p_t, p_zqs, wb_r1 + (size_t)r * D_MOD * D_MOD, nullptr,
           T_TOK, D_MOD, D_MOD, D_MOD, D_MOD, 0, D_MOD, 2);          // relu
      GEMM(p_zqs, p_t, wb_r2 + (size_t)r * D_MOD * D_MOD, nullptr,
           T_TOK, D_MOD, D_MOD, D_MOD, D_MOD, 0, D_MOD, 4);          // accumulate
    }

    add_k<<<GS_BLOCKS, 256, 0, stream>>>(p_xhat, p_zqs, T * D_MOD);
    sqerr_k<<<T_TOK, 256, 0, stream>>>(p_xhat, x, loss_out + (size_t)i * T);
  }

  // ---- decoder
  GEMM(p_h, p_xhat, wb_d1, dec_b1, T_TOK, 2 * D_MOD, D_MOD, D_MOD, D_MOD, 0, 2 * D_MOD, 1);
  colstats_k<<<2 * D_MOD, 256, 0, stream>>>(p_h, p_mean, p_var, T_TOK, 2 * D_MOD);
  bn_relu_k<<<GS_BLOCKS, 256, 0, stream>>>(p_h, p_mean, p_var, dbn1_g, dbn1_b,
                                           T * 2 * D_MOD, 2 * D_MOD);
  GEMM(out, p_h, wb_d2, dec_b2, T_TOK, D_MOD, 2 * D_MOD, 2 * D_MOD, 2 * D_MOD, 0, D_MOD, 1);
  colstats_k<<<D_MOD, 256, 0, stream>>>(out, p_mean, p_var, T_TOK, D_MOD);
  bn_relu_k<<<GS_BLOCKS, 256, 0, stream>>>(out, p_mean, p_var, dbn2_g, dbn2_b,
                                           T * D_MOD, D_MOD);
}
